// MeshGraphNet_85487029060210
// MI455X (gfx1250) — matrix operand order repaired and checked by reading, not compiled, not run on hardware
//
#include <hip/hip_runtime.h>
#include <hip/hip_bf16.h>
#include <stdint.h>
#include <stddef.h>

typedef __bf16 bf16;
typedef __attribute__((ext_vector_type(16))) __bf16 v16bf;
typedef __attribute__((ext_vector_type(8)))  __bf16 v8bf;
typedef __attribute__((ext_vector_type(8)))  float  v8f;

// ---------------------------------------------------------------------------
// Utility kernels
// ---------------------------------------------------------------------------
__global__ void k_zero_u32(uint32_t* __restrict__ p, size_t n) {
  size_t i = (size_t)blockIdx.x * blockDim.x + threadIdx.x;
  if (i < n) p[i] = 0u;
}

// Pack a row-major f32 weight W[Ksrc x Nout] into bf16 B-fragment layout:
// tile index [nb][kb], 32 lanes x 16 bf16 each; lane = (kin/16)*16 + (c%16),
// value j = kin%16 (K = kb*32 + (lane/16)*16 + j, col = nb*16 + lane%16).
__global__ void k_pack_w(const float* __restrict__ W, bf16* __restrict__ out,
                         int Ksrc, int Nout, int rowOff, int KP) {
  int idx = blockIdx.x * blockDim.x + threadIdx.x;
  int total = Ksrc * Nout;
  if (idx >= total) return;
  int r = idx / Nout, c = idx % Nout;
  int k = r + rowOff;
  int kb = k >> 5, kin = k & 31;
  int q = kin >> 4, j = kin & 15;
  int lane = q * 16 + (c & 15);
  int nb = c >> 4;
  int KB = KP >> 5;
  out[(((size_t)(nb * KB + kb)) * 32 + lane) * 16 + j] = (bf16)W[idx];
}

// ---------------------------------------------------------------------------
// Generic WMMA GEMM: C[M x Nout] = act(A[M x KP](bf16) @ Wp + bias [+ resid])
// Block: 256 threads = 8 waves (2 M-tiles x 4 N-tiles). Grid: (ceil(M/32), Nout/64)
// cmode: 0 = f32 out, 1 = bf16 out.
// ---------------------------------------------------------------------------
__global__ __launch_bounds__(256)
void k_gemm(const bf16* __restrict__ A, int M, int KP,
            const bf16* __restrict__ Wp, const float* __restrict__ bias,
            int Nout, void* __restrict__ Cout, int cmode,
            const bf16* __restrict__ resid, int relu) {
  const int tid  = threadIdx.x;
  const int wave = tid >> 5, lane = tid & 31;
  const int mrow = wave >> 2, ncol = wave & 3;
  const int mbase = blockIdx.x * 32 + mrow * 16;
  const int nb    = blockIdx.y * 4 + ncol;
  const int nbase = nb * 16;
  const int q  = lane >> 4;      // half-wave selector
  const int mr = lane & 15;      // fragment row / output column-in-tile
  int m = mbase + mr; if (m >= M) m = M - 1;   // clamp: safe loads, masked stores
  const bf16* __restrict__ arow = A + (size_t)m * KP;
  const int KB = KP >> 5;
  const bf16* __restrict__ bp = Wp + ((size_t)nb * KB * 32 + lane) * 16;

  v8f acc = {0.f, 0.f, 0.f, 0.f, 0.f, 0.f, 0.f, 0.f};
  for (int kb = 0; kb < KB; ++kb) {
    const int k0 = kb * 32 + q * 16;
    // A in contiguous k order (same as packed B): lanes 0-15 K{0..15}, lanes 16-31 K{16..31}
    v8bf a0 = *(const v8bf*)(arow + k0);
    v8bf a1 = *(const v8bf*)(arow + k0 + 8);
    v16bf av = __builtin_shufflevector(a0, a1,
                 0,1,2,3,4,5,6,7,8,9,10,11,12,13,14,15);
    v16bf bv = *(const v16bf*)(bp + (size_t)kb * 512);
    if (kb + 1 < KB) __builtin_prefetch(arow + k0 + 32, 0, 1);
    acc = __builtin_amdgcn_wmma_f32_16x16x32_bf16(
            false, av, false, bv, (short)0, acc, false, false);
  }

  const int col = nbase + mr;
  const float bb = bias ? bias[col] : 0.f;
  #pragma unroll
  for (int i = 0; i < 8; ++i) {
    int row = mbase + q * 8 + i;            // C layout: VGPR i -> M = i + q*8
    if (row >= M) continue;
    float v = acc[i] + bb;
    if (resid) v += (float)resid[(size_t)row * Nout + col];
    if (relu && v < 0.f) v = 0.f;
    if (cmode == 0) ((float*)Cout)[(size_t)row * Nout + col] = v;
    else            ((bf16*)Cout)[(size_t)row * Nout + col]  = (bf16)v;
  }
}

// ---------------------------------------------------------------------------
// LSTM pointwise gates: G[N x 512] (i,f,g,o) -> update c (f32), h (bf16), seq
// ---------------------------------------------------------------------------
__device__ __forceinline__ float sigmoidf_(float x) { return 1.f / (1.f + __expf(-x)); }

__global__ void k_lstm_gates(const float* __restrict__ G, float* __restrict__ c,
                             bf16* __restrict__ h, bf16* __restrict__ seqout,
                             int t, int N) {
  int idx = blockIdx.x * blockDim.x + threadIdx.x;
  if (idx >= N * 128) return;
  int n = idx >> 7, j = idx & 127;
  const float* g = G + (size_t)n * 512;
  float i_ = sigmoidf_(g[j]);
  float f_ = sigmoidf_(g[128 + j]);
  float gg = tanhf(g[256 + j]);
  float o_ = sigmoidf_(g[384 + j]);
  float cv = f_ * c[idx] + i_ * gg;
  float hv = o_ * tanhf(cv);
  c[idx] = cv;
  h[idx] = (bf16)hv;
  seqout[((size_t)n * 3 + t) * 128 + j] = (bf16)hv;
}

// ---------------------------------------------------------------------------
// Concat / pack kernels (build bf16 A matrices, zero-padded K)
// ---------------------------------------------------------------------------
__global__ void k_pack_lstm0(bf16* __restrict__ A, const float* __restrict__ x,
                             const bf16* __restrict__ h, int t, int N) {
  int idx = blockIdx.x * blockDim.x + threadIdx.x;
  if (idx >= N * 160) return;
  int n = idx / 160, c = idx % 160;
  bf16 v = (bf16)0.f;
  if (c < 12)        v = (bf16)x[((size_t)n * 12 + c) * 3 + t];
  else if (c < 140)  v = h[(size_t)n * 128 + (c - 12)];
  A[idx] = v;
}

__global__ void k_pack_lstm1(bf16* __restrict__ A, const bf16* __restrict__ seq,
                             const bf16* __restrict__ h, int t, int N) {
  int idx = blockIdx.x * blockDim.x + threadIdx.x;
  if (idx >= N * 256) return;
  int n = idx >> 8, c = idx & 255;
  A[idx] = (c < 128) ? seq[((size_t)n * 3 + t) * 128 + c]
                     : h[(size_t)n * 128 + (c - 128)];
}

__global__ void k_pack_temp(bf16* __restrict__ A, const bf16* __restrict__ h,
                            const float* __restrict__ mass,
                            const float* __restrict__ x, int N) {
  int idx = blockIdx.x * blockDim.x + threadIdx.x;
  if (idx >= N * 160) return;
  int n = idx / 160, c = idx % 160;
  bf16 v = (bf16)0.f;
  if (c < 128)       v = h[(size_t)n * 128 + c];
  else if (c == 128) v = (bf16)mass[n];
  else if (c < 132)  v = (bf16)x[((size_t)n * 12 + (c - 129)) * 3 + 2]; // coords
  A[idx] = v;
}

__global__ void k_pack_coords(bf16* __restrict__ A, const float* __restrict__ x, int N) {
  int idx = blockIdx.x * blockDim.x + threadIdx.x;
  if (idx >= N * 32) return;
  int n = idx >> 5, c = idx & 31;
  A[idx] = (c < 3) ? (bf16)x[((size_t)n * 12 + c) * 3 + 2] : (bf16)0.f;
}

__global__ void k_pack_edge_enc(bf16* __restrict__ A, const float* __restrict__ ea,
                                const float* __restrict__ memb, int E) {
  int idx = blockIdx.x * blockDim.x + threadIdx.x;
  if (idx >= E * 32) return;
  int e = idx >> 5, c = idx & 31;
  bf16 v = (bf16)0.f;
  if (c < 4) { int mid = (int)ea[(size_t)e * 4]; v = (bf16)memb[mid * 4 + c]; }
  else if (c < 7) v = (bf16)ea[(size_t)e * 4 + 1 + (c - 4)];
  A[idx] = v;
}

__global__ void k_pack_topo_edge(bf16* __restrict__ A, const bf16* __restrict__ m,
                                 const bf16* __restrict__ node,
                                 const int* __restrict__ ei, int E) {
  int idx = blockIdx.x * blockDim.x + threadIdx.x;
  if (idx >= E * 48) return;
  int e = idx / 48, c0 = (idx % 48) * 8;
  const bf16* s;
  if (c0 < 128)      s = m    + (size_t)e * 128 + c0;
  else if (c0 < 256) s = node + (size_t)ei[e]     * 128 + (c0 - 128);
  else               s = node + (size_t)ei[E + e] * 128 + (c0 - 256);
  *(v8bf*)(A + (size_t)e * 384 + c0) = *(const v8bf*)s;
}

__global__ void k_pack_rad_edge(bf16* __restrict__ A, const float* __restrict__ rattr,
                                const bf16* __restrict__ node,
                                const int* __restrict__ ei, int E) {
  int idx = blockIdx.x * blockDim.x + threadIdx.x;
  if (idx >= E * 288) return;
  int e = idx / 288, c = idx % 288;
  bf16 v = (bf16)0.f;
  if (c == 0)        v = (bf16)rattr[e];
  else if (c < 129)  v = node[(size_t)ei[e]     * 128 + (c - 1)];
  else if (c < 257)  v = node[(size_t)ei[E + e] * 128 + (c - 129)];
  A[(size_t)e * 288 + c] = v;
}

__global__ void k_pack_node_agg(bf16* __restrict__ A, const bf16* __restrict__ node,
                                const float* __restrict__ agg, int N) {
  int idx = blockIdx.x * blockDim.x + threadIdx.x;
  if (idx >= N * 32) return;
  int n = idx >> 5, c0 = (idx & 31) * 8;
  bf16* dst = A + (size_t)n * 256 + c0;
  if (c0 < 128) {
    *(v8bf*)dst = *(const v8bf*)(node + (size_t)n * 128 + c0);
  } else {
    const float* s = agg + (size_t)n * 128 + (c0 - 128);
    v8bf r;
    #pragma unroll
    for (int j = 0; j < 8; ++j) r[j] = (bf16)s[j];
    *(v8bf*)dst = r;
  }
}

__global__ void k_pack_cat2(bf16* __restrict__ A, const bf16* __restrict__ a,
                            const bf16* __restrict__ b, int N) {
  int idx = blockIdx.x * blockDim.x + threadIdx.x;
  if (idx >= N * 32) return;
  int n = idx >> 5, c0 = (idx & 31) * 8;
  const bf16* s = (c0 < 128) ? (a + (size_t)n * 128 + c0)
                             : (b + (size_t)n * 128 + (c0 - 128));
  *(v8bf*)(A + (size_t)n * 256 + c0) = *(const v8bf*)s;
}

__global__ void k_scatter_agg(float* __restrict__ agg, const bf16* __restrict__ m,
                              const int* __restrict__ dst, int E) {
  size_t idx = (size_t)blockIdx.x * blockDim.x + threadIdx.x;
  if (idx >= (size_t)E * 128) return;
  int e = (int)(idx >> 7), j = (int)(idx & 127);
  atomicAdd(&agg[(size_t)dst[e] * 128 + j], (float)m[idx]);
}

__global__ void k_decoder_final(const bf16* __restrict__ hid, const float* __restrict__ W,
                                const float* __restrict__ b, float* __restrict__ out, int N) {
  int idx = blockIdx.x * blockDim.x + threadIdx.x;
  if (idx >= N * 3) return;
  int n = idx / 3, j = idx % 3;
  float s = b[j];
  const bf16* hrow = hid + (size_t)n * 128;
  #pragma unroll 8
  for (int k = 0; k < 128; ++k) s += (float)hrow[k] * W[k * 3 + j];
  out[idx] = s;
}

// ---------------------------------------------------------------------------
// Host orchestration
// ---------------------------------------------------------------------------
extern "C" void kernel_launch(void* const* d_in, const int* in_sizes, int n_in,
                              void* d_out, int out_size, void* d_ws, size_t ws_size,
                              hipStream_t stream) {
  (void)in_sizes; (void)n_in; (void)out_size;
  const int N = 50000, E = 300000;

  // ---- workspace layout (bytes, all 256-aligned) ----
  const size_t off_A    = 0;            // 230,400,000 (E*384 bf16)  [LSTM A uses first 25.6MB]
  const size_t off_G    = 116000000;    // 102,400,000 (N*512 f32)   inside A region, LSTM-only
  const size_t off_m    = 230400000;    //  76,800,000 (E*128 bf16)  [hosts LSTM seq bufs first]
  const size_t off_seqA = off_m;
  const size_t off_seqB = off_m + 38400000;
  const size_t off_hid  = 307200000;    //  76,800,000 (E*128 bf16)
  const size_t off_hT   = 384000000;    //  12,800,000
  const size_t off_hR   = 396800000;    //  12,800,000
  const size_t off_agg  = 409600000;    //  25,600,000 (N*128 f32)
  const size_t off_h    = 435200000;    //  12,800,000 (lstm h / h_last / add_passage out)
  const size_t off_c    = 448000000;    //  25,600,000 (N*128 f32)
  const size_t off_W    = 473600000;    //   1,376,256 packed bf16 weights
  const size_t W_ELEMS  = 688128;
  if (ws_size < off_W + W_ELEMS * 2) return;  // not enough scratch: bail safely

  char* ws = (char*)d_ws;
  bf16*  Abuf = (bf16*)(ws + off_A);
  float* G    = (float*)(ws + off_G);
  bf16*  mbuf = (bf16*)(ws + off_m);
  bf16*  seqA = (bf16*)(ws + off_seqA);
  bf16*  seqB = (bf16*)(ws + off_seqB);
  bf16*  hid  = (bf16*)(ws + off_hid);
  bf16*  hT   = (bf16*)(ws + off_hT);
  bf16*  hR   = (bf16*)(ws + off_hR);
  float* agg  = (float*)(ws + off_agg);
  bf16*  h    = (bf16*)(ws + off_h);
  float* cbuf = (float*)(ws + off_c);
  bf16*  wb   = (bf16*)(ws + off_W);

  // ---- inputs (JAX pytree flatten: sorted dict keys; 'W'<'b', 'Whh'<'Wih'<'b') ----
  const float* x     = (const float*)d_in[0];
  const float* mass  = (const float*)d_in[1];
  const float* ea    = (const float*)d_in[2];
  const int*   ei    = (const int*)d_in[3];
  const int*   rei   = (const int*)d_in[4];
  const float* rattr = (const float*)d_in[5];
  const float* memb  = (const float*)d_in[25];
  auto B = [&](int i) { return (const float*)d_in[i]; };

  // ---- launch helpers ----
  auto zero = [&](void* p, size_t bytes) {
    size_t words = bytes >> 2;
    k_zero_u32<<<dim3((unsigned)((words + 255) / 256)), 256, 0, stream>>>((uint32_t*)p, words);
  };
  auto packw = [&](int widx, bf16* dst, int Ksrc, int Nout, int rowOff, int KP) {
    int tot = Ksrc * Nout;
    k_pack_w<<<dim3((tot + 255) / 256), 256, 0, stream>>>((const float*)d_in[widx], dst, Ksrc, Nout, rowOff, KP);
  };
  auto gemm = [&](const bf16* A, int M, int KP, const bf16* Wp, const float* bias,
                  int Nout, void* C, int cmode, const bf16* resid, int relu) {
    dim3 g((M + 31) / 32, Nout / 64);
    k_gemm<<<g, 256, 0, stream>>>(A, M, KP, Wp, bias, Nout, C, cmode, resid, relu);
  };
  auto g1 = [&](size_t n) { return dim3((unsigned)((n + 255) / 256)); };

  // ---- packed weight carve-out ----
  size_t wo = 0;
  auto walloc = [&](size_t e) { bf16* p = wb + wo; wo += e; return p; };
  bf16* wp_lstm0 = walloc(160 * 512);
  bf16* wp_lstm1 = walloc(256 * 512);
  bf16* wp_lstm2 = walloc(256 * 512);
  bf16* wp_temp0 = walloc(160 * 128); bf16* wp_temp1 = walloc(128 * 128);
  bf16* wp_nrad0 = walloc(32 * 128);  bf16* wp_nrad1 = walloc(128 * 128);
  bf16* wp_eenc0 = walloc(32 * 128);  bf16* wp_eenc1 = walloc(128 * 128);
  bf16* wp_te0   = walloc(384 * 128); bf16* wp_te1   = walloc(128 * 128);
  bf16* wp_tn0   = walloc(256 * 128); bf16* wp_tn1   = walloc(128 * 128);
  bf16* wp_re0   = walloc(288 * 128); bf16* wp_re1   = walloc(128 * 128);
  bf16* wp_rn0   = walloc(256 * 128); bf16* wp_rn1   = walloc(128 * 128);
  bf16* wp_ap    = walloc(256 * 128);
  bf16* wp_dec0  = walloc(128 * 128);

  zero(wb, W_ELEMS * 2);
  packw(17, wp_lstm0, 12, 512, 0, 160);   packw(16, wp_lstm0, 128, 512, 12, 160);
  packw(20, wp_lstm1, 128, 512, 0, 256);  packw(19, wp_lstm1, 128, 512, 128, 256);
  packw(23, wp_lstm2, 128, 512, 0, 256);  packw(22, wp_lstm2, 128, 512, 128, 256);
  packw(38, wp_temp0, 132, 128, 0, 160);  packw(40, wp_temp1, 128, 128, 0, 128);
  packw(26, wp_nrad0, 3, 128, 0, 32);     packw(28, wp_nrad1, 128, 128, 0, 128);
  packw(12, wp_eenc0, 7, 128, 0, 32);     packw(14, wp_eenc1, 128, 128, 0, 128);
  packw(42, wp_te0, 384, 128, 0, 384);    packw(44, wp_te1, 128, 128, 0, 128);
  packw(46, wp_tn0, 256, 128, 0, 256);    packw(48, wp_tn1, 128, 128, 0, 128);
  packw(30, wp_re0, 257, 128, 0, 288);    packw(32, wp_re1, 128, 128, 0, 128);
  packw(34, wp_rn0, 256, 128, 0, 256);    packw(36, wp_rn1, 128, 128, 0, 128);
  packw(6,  wp_ap, 256, 128, 0, 256);
  packw(8,  wp_dec0, 128, 128, 0, 128);

  // ---- 3-layer LSTM over T=3 ----
  bf16* wp_lstm[3] = {wp_lstm0, wp_lstm1, wp_lstm2};
  const float* bL[3] = {B(18), B(21), B(24)};
  int kpL[3] = {160, 256, 256};
  bf16* sin_ = seqB; bf16* sout_ = seqA;
  for (int l = 0; l < 3; ++l) {
    if (l > 0) { bf16* t = sin_; sin_ = sout_; sout_ = t; }
    zero(h, (size_t)N * 128 * 2);
    zero(cbuf, (size_t)N * 128 * 4);
    for (int t = 0; t < 3; ++t) {
      if (l == 0) k_pack_lstm0<<<g1((size_t)N * 160), 256, 0, stream>>>(Abuf, x, h, t, N);
      else        k_pack_lstm1<<<g1((size_t)N * 256), 256, 0, stream>>>(Abuf, sin_, h, t, N);
      gemm(Abuf, N, kpL[l], wp_lstm[l], bL[l], 512, G, 0, nullptr, 0);
      k_lstm_gates<<<g1((size_t)N * 128), 256, 0, stream>>>(G, cbuf, h, sout_, t, N);
    }
  }
  // h now holds h_last (N x 128 bf16)

  // ---- temp_fc: [h | mass | coords] -> hT ----
  k_pack_temp<<<g1((size_t)N * 160), 256, 0, stream>>>(Abuf, h, mass, x, N);
  gemm(Abuf, N, 160, wp_temp0, B(39), 128, hid, 1, nullptr, 1);
  gemm(hid,  N, 128, wp_temp1, B(41), 128, hT,  1, nullptr, 0);

  // ---- node_radius: coords -> hR ----
  k_pack_coords<<<g1((size_t)N * 32), 256, 0, stream>>>(Abuf, x, N);
  gemm(Abuf, N, 32,  wp_nrad0, B(27), 128, hid, 1, nullptr, 1);
  gemm(hid,  N, 128, wp_nrad1, B(29), 128, hR,  1, nullptr, 0);

  // ---- edge_enc: [mat_emb | num] -> m ----
  k_pack_edge_enc<<<g1((size_t)E * 32), 256, 0, stream>>>(Abuf, ea, memb, E);
  gemm(Abuf, E, 32,  wp_eenc0, B(13), 128, hid,  1, nullptr, 1);
  gemm(hid,  E, 128, wp_eenc1, B(15), 128, mbuf, 1, nullptr, 0);

  // ---- 4x topo GNN layers (edge residual) ----
  for (int l = 0; l < 4; ++l) {
    k_pack_topo_edge<<<g1((size_t)E * 48), 256, 0, stream>>>(Abuf, mbuf, hT, ei, E);
    gemm(Abuf, E, 384, wp_te0, B(43), 128, hid,  1, nullptr, 1);
    gemm(hid,  E, 128, wp_te1, B(45), 128, mbuf, 1, mbuf, 0);   // m = mlp + m
    zero(agg, (size_t)N * 128 * 4);
    k_scatter_agg<<<g1((size_t)E * 128), 256, 0, stream>>>(agg, mbuf, ei + E, E);
    k_pack_node_agg<<<g1((size_t)N * 32), 256, 0, stream>>>(Abuf, hT, agg, N);
    gemm(Abuf, N, 256, wp_tn0, B(47), 128, hid, 1, nullptr, 1);
    gemm(hid,  N, 128, wp_tn1, B(49), 128, hT,  1, hT, 0);      // node = node + mlp
  }

  // ---- radius GNN (no edge residual) ----
  k_pack_rad_edge<<<g1((size_t)E * 288), 256, 0, stream>>>(Abuf, rattr, hR, rei, E);
  gemm(Abuf, E, 288, wp_re0, B(31), 128, hid,  1, nullptr, 1);
  gemm(hid,  E, 128, wp_re1, B(33), 128, mbuf, 1, nullptr, 0);
  zero(agg, (size_t)N * 128 * 4);
  k_scatter_agg<<<g1((size_t)E * 128), 256, 0, stream>>>(agg, mbuf, rei + E, E);
  k_pack_node_agg<<<g1((size_t)N * 32), 256, 0, stream>>>(Abuf, hR, agg, N);
  gemm(Abuf, N, 256, wp_rn0, B(35), 128, hid, 1, nullptr, 1);
  gemm(hid,  N, 128, wp_rn1, B(37), 128, hR,  1, hR, 0);

  // ---- add_passage (single layer, no relu): [hT | hR] -> hAP (reuse h buf) ----
  k_pack_cat2<<<g1((size_t)N * 32), 256, 0, stream>>>(Abuf, hT, hR, N);
  bf16* hAP = h;
  gemm(Abuf, N, 256, wp_ap, B(7), 128, hAP, 1, nullptr, 0);

  // ---- decoder: 128 -> 128 (relu) -> 3 ----
  gemm(hAP, N, 128, wp_dec0, B(9), 128, hid, 1, nullptr, 1);
  k_decoder_final<<<g1((size_t)N * 3), 256, 0, stream>>>(hid, B(10), B(11), (float*)d_out, N);
}
